// Attention_54898271978129
// MI455X (gfx1250) — matrix operand order repaired and checked by reading, not compiled, not run on hardware
//
#include <hip/hip_runtime.h>
#include <hip/hip_bf16.h>

// ---------------------------------------------------------------------------
// Talking-heads attention for MI455X (gfx1250, wave32, WMMA 16x16x32 bf16).
//
// Pipeline (all scratch in d_ws):
//   k0: transpose+convert Wqkv -> bf16 col-major        (WqkvT)
//   k1: transpose+convert Wout -> bf16 col-major        (WoutT)
//   k2: RMSNorm -> xn bf16, gates = sigmoid(xn@Wgate+b) (xn_bf, gates)
//   k3: QKV GEMM (WMMA bf16): q*sqrt(D) -> q_bf[b,h,n,d], k_bf[b,h,n,d],
//       v transposed -> vt_bf[b,h,d,n] (d-major so PV B-tiles load contiguous)
//   k4: fused two-pass flash attention with pre/post talking-heads mixing.
//       16 waves/block = 16 channels of one batch-group; channel mixes go
//       through padded LDS in fp32; QK^T and PV use v_wmma_f32_16x16x32_bf16.
//   k5: output GEMM (WMMA bf16) -> d_out fp32.
// ---------------------------------------------------------------------------

typedef __attribute__((ext_vector_type(16))) __bf16 v16bf;
typedef __attribute__((ext_vector_type(8)))  __bf16 v8bf;
typedef __attribute__((ext_vector_type(8)))  float  v8f;

#define WMMA_BF16(A, B, C) \
  __builtin_amdgcn_wmma_f32_16x16x32_bf16(false, (A), false, (B), (short)0, (C), false, false)

constexpr int B_   = 4;
constexpr int NSEQ = 2048;
constexpr int DIMM = 512;
constexpr int H_   = 8;
constexpr int DH   = 64;
constexpr int S_   = 2;
constexpr int NCH  = 16;        // S*H talking-head channels per batch group
constexpr int ROWS = B_ * NSEQ; // 8192

// workspace layout (bytes); every section is a multiple of 256B
constexpr size_t SZ_QKV_E = (size_t)B_ * H_ * NSEQ * DH;   // 4,194,304 elems
constexpr size_t OFF_Q  = 0;
constexpr size_t OFF_K  = OFF_Q  + SZ_QKV_E * 2;
constexpr size_t OFF_VT = OFF_K  + SZ_QKV_E * 2;
constexpr size_t OFF_XN = OFF_VT + SZ_QKV_E * 2;
constexpr size_t OFF_AO = OFF_XN + (size_t)ROWS * DIMM * 2;
constexpr size_t OFF_WQ = OFF_AO + (size_t)ROWS * DIMM * 2;
constexpr size_t OFF_WO = OFF_WQ + (size_t)1536 * 512 * 2;
constexpr size_t OFF_G  = OFF_WO + (size_t)512 * 512 * 2;
// total ~42.5 MB

// ---- A-operand loader: 16x32 bf16 tile, contiguous k order (k = 16*hi + s) ------
// lanes 0-15 : row M=lane,    K = {k0+0..15}
// lanes 16-31: row M=lane-16, K = {k0+16..31}
__device__ inline v16bf load_a_tile(const __bf16* __restrict__ base, int stride,
                                    int mr, int hi, int k0) {
  const __bf16* p = base + (size_t)mr * stride + k0 + 16 * hi;
  v8bf lo = *(const v8bf*)p;
  v8bf hh = *(const v8bf*)(p + 8);
  v16bf r;
#pragma unroll
  for (int t = 0; t < 8; ++t) { r[t] = lo[t]; r[t + 8] = hh[t]; }
  return r;
}

// ---------------- k0/k1: weight transpose + bf16 convert --------------------
__global__ void convert_wqkv_kernel(const float* __restrict__ W, __bf16* __restrict__ WT) {
  int idx = blockIdx.x * 256 + threadIdx.x;          // over 1536*512
  if (idx >= 1536 * 512) return;
  int col = idx >> 9, k = idx & 511;
  WT[idx] = (__bf16)W[(size_t)k * 1536 + col];       // WT[col][k]
}

__global__ void convert_wout_kernel(const float* __restrict__ W, __bf16* __restrict__ WT) {
  int idx = blockIdx.x * 256 + threadIdx.x;          // over 512*512
  if (idx >= 512 * 512) return;
  int col = idx >> 9, k = idx & 511;
  WT[idx] = (__bf16)W[(size_t)k * 512 + col];
}

// ---------------- k2: RMSNorm + gates ---------------------------------------
__global__ __launch_bounds__(256)
void prep_xn_kernel(const float* __restrict__ x, const float* __restrict__ g,
                    const float* __restrict__ Wgate, const float* __restrict__ bgate,
                    __bf16* __restrict__ xn_bf, float* __restrict__ gates) {
  int w = threadIdx.x >> 5, lane = threadIdx.x & 31;
  int row = blockIdx.x * 8 + w;                      // 0..8191
  const float* xr = x + (size_t)row * DIMM;
  float xv[16]; float ss = 0.f;
#pragma unroll
  for (int t = 0; t < 16; ++t) { float v = xr[t * 32 + lane]; xv[t] = v; ss += v * v; }
#pragma unroll
  for (int s = 1; s < 32; s <<= 1) ss += __shfl_xor(ss, s, 32);
  float rms = rsqrtf(ss * (1.0f / DIMM) + 1e-5f);
  float gp[8];
#pragma unroll
  for (int hh = 0; hh < 8; ++hh) gp[hh] = 0.f;
  __bf16* xo = xn_bf + (size_t)row * DIMM;
#pragma unroll
  for (int t = 0; t < 16; ++t) {
    int d = t * 32 + lane;
    float xn = xv[t] * rms * g[d];
    xo[d] = (__bf16)xn;
    const float* wg = Wgate + (size_t)d * 8;
#pragma unroll
    for (int hh = 0; hh < 8; ++hh) gp[hh] += xn * wg[hh];
  }
#pragma unroll
  for (int hh = 0; hh < 8; ++hh)
#pragma unroll
    for (int s = 1; s < 32; s <<= 1) gp[hh] += __shfl_xor(gp[hh], s, 32);
  if (lane < 8) {
    float z = gp[lane] + bgate[lane];
    gates[(size_t)row * H_ + lane] = 1.0f / (1.0f + __expf(-z));
  }
}

// ---------------- k3: QKV GEMM (WMMA) + scatter -----------------------------
__global__ __launch_bounds__(256)
void qkv_gemm_kernel(const __bf16* __restrict__ xn, const __bf16* __restrict__ WT,
                     __bf16* __restrict__ qb, __bf16* __restrict__ kb,
                     __bf16* __restrict__ vtb) {
  int w = threadIdx.x >> 5, lane = threadIdx.x & 31;
  int ct = blockIdx.x * 8 + w;   // 0..95  (1536/16 column tiles)
  int rt = blockIdx.y;           // 0..511 (8192/16 row tiles)
  int i0 = rt * 16, c0 = ct * 16;
  int mr = lane & 15, hi = lane >> 4;
  v8f acc = {};
  const __bf16* arow = xn + (size_t)i0 * DIMM;
  const __bf16* brow = WT + (size_t)c0 * DIMM;
#pragma unroll 4
  for (int k0 = 0; k0 < DIMM; k0 += 32) {
    v16bf a = load_a_tile(arow, DIMM, mr, hi, k0);
    v16bf b = *(const v16bf*)(brow + (size_t)mr * DIMM + k0 + 16 * hi);
    acc = WMMA_BF16(a, b, acc);
  }
  int col = c0 + mr;
  int three = col >> 9, hcol = (col >> 6) & 7, d = col & 63;
#pragma unroll
  for (int r = 0; r < 8; ++r) {
    int i = i0 + r + 8 * hi;                 // global row in [0,8192)
    int bb = i >> 11, nn = i & 2047;
    float v = acc[r];
    if (three == 0) {
      qb[(((size_t)(bb * H_ + hcol)) * NSEQ + nn) * DH + d] = (__bf16)(v * 8.0f); // *sqrt(D)
    } else if (three == 1) {
      kb[(((size_t)(bb * H_ + hcol)) * NSEQ + nn) * DH + d] = (__bf16)v;
    } else {
      vtb[(((size_t)(bb * H_ + hcol)) * DH + d) * NSEQ + nn] = (__bf16)v;         // d-major
    }
  }
}

// ---------------- k4: fused talking-heads flash attention -------------------
constexpr int EX_ROW = 34;                 // fp32 exchange row stride (pad)
constexpr int EX_CH  = 16 * EX_ROW + 1;    // 545 floats per channel (odd -> bank spread)
constexpr int PM_ROW = 34;                 // bf16 staging row stride
constexpr int PM_CH  = 16 * PM_ROW;        // per-wave private pm tile

__global__ __launch_bounds__(512)
void attn_core_kernel(const __bf16* __restrict__ qb, const __bf16* __restrict__ kb,
                      const __bf16* __restrict__ vtb,
                      const float* __restrict__ Wpre, const float* __restrict__ Wpost,
                      const unsigned char* __restrict__ mask,
                      const float* __restrict__ gates, __bf16* __restrict__ ao) {
  __shared__ float  s_ex[NCH * EX_CH];     // fp32 channel exchange (~34.9 KB)
  __shared__ __bf16 s_pm[NCH * PM_CH];     // per-wave pm staging   (~17.4 KB)

  int w = threadIdx.x >> 5, lane = threadIdx.x & 31;   // w = channel 0..15
  int gb = blockIdx.x;                                  // batch group 0..1
  int i0 = blockIdx.y * 16;                             // query tile
  int b = gb * S_ + (w >> 3), h = w & 7;
  int mr = lane & 15, hi = lane >> 4;

  const __bf16* qbase = qb  + (((size_t)(b * H_ + h)) * NSEQ + i0) * DH;
  const __bf16* kbase = kb  +  ((size_t)(b * H_ + h)) * NSEQ * DH;
  const __bf16* vbase = vtb +  ((size_t)(b * H_ + h)) * DH * NSEQ;
  const unsigned char* mrow = mask + (size_t)b * NSEQ;

  float wpre_r[16], wpost_r[16];
#pragma unroll
  for (int c2 = 0; c2 < 16; ++c2) {
    wpre_r[c2]  = Wpre[w * 16 + c2];
    wpost_r[c2] = Wpost[w * 16 + c2];
  }

  v16bf qa0 = load_a_tile(qbase, DH, mr, hi, 0);
  v16bf qa1 = load_a_tile(qbase, DH, mr, hi, 32);

  float m[8], l[8];
#pragma unroll
  for (int r = 0; r < 8; ++r) { m[r] = -3.0e38f; l[r] = 0.f; }
  int nChunks = (i0 + 15) / 32 + 1;        // causal: keys up to i0+15
  float* myex = s_ex + w * EX_CH;
  int jg0base, jg1base;

  // ---------------- Pass A: softmax statistics (online) ----------------
  for (int ch = 0; ch < nChunks; ++ch) {
    int j0 = ch * 32;
    // prefetch next K chunk
    {
      int jp = j0 + 32 + mr; if (jp > NSEQ - 1) jp = NSEQ - 1;
      __builtin_prefetch(kbase + (size_t)jp * DH, 0, 0);
    }
    v8f s0 = {}, s1 = {};
    {
      const __bf16* k0p = kbase + (size_t)(j0 + mr) * DH + 16 * hi;
      s0 = WMMA_BF16(qa0, *(const v16bf*)k0p,        s0);
      s0 = WMMA_BF16(qa1, *(const v16bf*)(k0p + 32), s0);
      const __bf16* k1p = kbase + (size_t)(j0 + 16 + mr) * DH + 16 * hi;
      s1 = WMMA_BF16(qa0, *(const v16bf*)k1p,        s1);
      s1 = WMMA_BF16(qa1, *(const v16bf*)(k1p + 32), s1);
    }
#pragma unroll
    for (int r = 0; r < 8; ++r) {          // write raw sim to exchange
      myex[(r + 8 * hi) * EX_ROW + mr]      = s0[r];
      myex[(r + 8 * hi) * EX_ROW + 16 + mr] = s1[r];
    }
    __syncthreads();
    // pre talking-heads mix (fp32, read all 16 channels)
    float mx0[8], mx1[8];
#pragma unroll
    for (int r = 0; r < 8; ++r) {
      int ii = r + 8 * hi;
      float a0 = 0.f, a1 = 0.f;
#pragma unroll
      for (int c2 = 0; c2 < 16; ++c2) {
        const float* p = s_ex + c2 * EX_CH + ii * EX_ROW;
        a0 += wpre_r[c2] * p[mr];
        a1 += wpre_r[c2] * p[16 + mr];
      }
      mx0[r] = a0; mx1[r] = a1;
    }
    jg0base = j0 + mr; jg1base = jg0base + 16;
    bool k0ok = mrow[jg0base] != 0, k1ok = mrow[jg1base] != 0;
#pragma unroll
    for (int r = 0; r < 8; ++r) {
      int ig = i0 + r + 8 * hi;
      float v0 = (k0ok && jg0base <= ig) ? mx0[r] : -3.0e38f;
      float v1 = (k1ok && jg1base <= ig) ? mx1[r] : -3.0e38f;
      float t = fmaxf(v0, v1);
#pragma unroll
      for (int s = 1; s < 16; s <<= 1) t = fmaxf(t, __shfl_xor(t, s, 32));
      float mn = fmaxf(m[r], t);
      float su = __expf(v0 - mn) + __expf(v1 - mn);
#pragma unroll
      for (int s = 1; s < 16; s <<= 1) su += __shfl_xor(su, s, 32);
      l[r] = l[r] * __expf(m[r] - mn) + su;
      m[r] = mn;
    }
    __syncthreads();
  }

  float invl[8];
#pragma unroll
  for (int r = 0; r < 8; ++r) invl[r] = 1.0f / l[r];

  // ---------------- Pass B: probabilities, post-mix, PV ----------------
  v8f oacc[4] = {};
  __bf16* mypm = s_pm + w * PM_CH;
  for (int ch = 0; ch < nChunks; ++ch) {
    int j0 = ch * 32;
    v8f s0 = {}, s1 = {};
    {
      const __bf16* k0p = kbase + (size_t)(j0 + mr) * DH + 16 * hi;
      s0 = WMMA_BF16(qa0, *(const v16bf*)k0p,        s0);
      s0 = WMMA_BF16(qa1, *(const v16bf*)(k0p + 32), s0);
      const __bf16* k1p = kbase + (size_t)(j0 + 16 + mr) * DH + 16 * hi;
      s1 = WMMA_BF16(qa0, *(const v16bf*)k1p,        s1);
      s1 = WMMA_BF16(qa1, *(const v16bf*)(k1p + 32), s1);
    }
#pragma unroll
    for (int r = 0; r < 8; ++r) {
      myex[(r + 8 * hi) * EX_ROW + mr]      = s0[r];
      myex[(r + 8 * hi) * EX_ROW + 16 + mr] = s1[r];
    }
    __syncthreads();
    float p0[8], p1[8];
    {
      int jg0 = j0 + mr, jg1 = jg0 + 16;
      bool k0ok = mrow[jg0] != 0, k1ok = mrow[jg1] != 0;
#pragma unroll
      for (int r = 0; r < 8; ++r) {
        int ii = r + 8 * hi;
        float a0 = 0.f, a1 = 0.f;
#pragma unroll
        for (int c2 = 0; c2 < 16; ++c2) {
          const float* p = s_ex + c2 * EX_CH + ii * EX_ROW;
          a0 += wpre_r[c2] * p[mr];
          a1 += wpre_r[c2] * p[16 + mr];
        }
        int ig = i0 + ii;
        float v0 = (k0ok && jg0 <= ig) ? a0 : -3.0e38f;
        float v1 = (k1ok && jg1 <= ig) ? a1 : -3.0e38f;
        p0[r] = __expf(v0 - m[r]) * invl[r];
        p1[r] = __expf(v1 - m[r]) * invl[r];
      }
    }
    __syncthreads();                       // premix reads done -> reuse s_ex
#pragma unroll
    for (int r = 0; r < 8; ++r) {          // publish normalized probs
      myex[(r + 8 * hi) * EX_ROW + mr]      = p0[r];
      myex[(r + 8 * hi) * EX_ROW + 16 + mr] = p1[r];
    }
    __syncthreads();
    // post talking-heads mix -> pm (fp32), stage wave-private as bf16
#pragma unroll
    for (int r = 0; r < 8; ++r) {
      int ii = r + 8 * hi;
      float a0 = 0.f, a1 = 0.f;
#pragma unroll
      for (int c2 = 0; c2 < 16; ++c2) {
        const float* p = s_ex + c2 * EX_CH + ii * EX_ROW;
        a0 += wpost_r[c2] * p[mr];
        a1 += wpost_r[c2] * p[16 + mr];
      }
      mypm[ii * PM_ROW + mr]      = (__bf16)a0;
      mypm[ii * PM_ROW + 16 + mr] = (__bf16)a1;
    }
    __syncthreads();                       // s_ex free for next chunk; pm visible
    // reload pm tile in A layout (wave-private LDS region)
    v16bf pa;
    {
      const __bf16* mp = mypm + mr * PM_ROW;
#pragma unroll
      for (int t = 0; t < 8; ++t) {
        pa[t]     = mp[16 * hi + t];
        pa[t + 8] = mp[16 * hi + 8 + t];
      }
    }
    // PV: out[i, d] += pm[i, j(32)] * v[j, d] for four 16-wide d tiles
#pragma unroll
    for (int dt = 0; dt < 4; ++dt) {
      v16bf vb = *(const v16bf*)(vbase + (size_t)(dt * 16 + mr) * NSEQ + j0 + 16 * hi);
      oacc[dt] = WMMA_BF16(pa, vb, oacc[dt]);
    }
  }

  // epilogue: gate, store gated output as bf16 rows (b, n, h*64+d)
#pragma unroll
  for (int r = 0; r < 8; ++r) {
    int ii = r + 8 * hi;
    float gate = gates[((size_t)b * NSEQ + i0 + ii) * H_ + h];
#pragma unroll
    for (int dt = 0; dt < 4; ++dt) {
      int d = dt * 16 + mr;
      ao[((size_t)b * NSEQ + i0 + ii) * DIMM + h * DH + d] = (__bf16)(oacc[dt][r] * gate);
    }
  }
}

// ---------------- k5: output GEMM (WMMA) ------------------------------------
__global__ __launch_bounds__(256)
void out_gemm_kernel(const __bf16* __restrict__ ao, const __bf16* __restrict__ WT,
                     float* __restrict__ out) {
  int w = threadIdx.x >> 5, lane = threadIdx.x & 31;
  int ct = blockIdx.x * 8 + w;   // 0..31
  int rt = blockIdx.y;           // 0..511
  int i0 = rt * 16, c0 = ct * 16;
  int mr = lane & 15, hi = lane >> 4;
  v8f acc = {};
  const __bf16* arow = ao + (size_t)i0 * DIMM;
  const __bf16* brow = WT + (size_t)c0 * DIMM;
#pragma unroll 4
  for (int k0 = 0; k0 < DIMM; k0 += 32) {
    v16bf a = load_a_tile(arow, DIMM, mr, hi, k0);
    v16bf b = *(const v16bf*)(brow + (size_t)mr * DIMM + k0 + 16 * hi);
    acc = WMMA_BF16(a, b, acc);
  }
#pragma unroll
  for (int r = 0; r < 8; ++r)
    out[(size_t)(i0 + r + 8 * hi) * DIMM + c0 + mr] = acc[r];
}

// ---------------------------------------------------------------------------
extern "C" void kernel_launch(void* const* d_in, const int* in_sizes, int n_in,
                              void* d_out, int out_size, void* d_ws, size_t ws_size,
                              hipStream_t stream) {
  const float* x             = (const float*)d_in[0];
  const unsigned char* mask  = (const unsigned char*)d_in[1]; // jnp bool -> u8
  const float* g             = (const float*)d_in[2];
  const float* Wqkv          = (const float*)d_in[3];
  const float* Wgate         = (const float*)d_in[4];
  const float* bgate         = (const float*)d_in[5];
  const float* Wpre          = (const float*)d_in[6];
  const float* Wpost         = (const float*)d_in[7];
  const float* Wout          = (const float*)d_in[8];
  float* out                 = (float*)d_out;

  char* ws = (char*)d_ws;
  __bf16* q_bf   = (__bf16*)(ws + OFF_Q);
  __bf16* k_bf   = (__bf16*)(ws + OFF_K);
  __bf16* vt_bf  = (__bf16*)(ws + OFF_VT);
  __bf16* xn_bf  = (__bf16*)(ws + OFF_XN);
  __bf16* ao_bf  = (__bf16*)(ws + OFF_AO);
  __bf16* wqkvT  = (__bf16*)(ws + OFF_WQ);
  __bf16* woutT  = (__bf16*)(ws + OFF_WO);
  float*  gates  = (float*)(ws + OFF_G);

  convert_wqkv_kernel<<<3072, 256, 0, stream>>>(Wqkv, wqkvT);
  convert_wout_kernel<<<1024, 256, 0, stream>>>(Wout, woutT);
  prep_xn_kernel<<<ROWS / 8, 256, 0, stream>>>(x, g, Wgate, bgate, xn_bf, gates);
  qkv_gemm_kernel<<<dim3(12, ROWS / 16), 256, 0, stream>>>(xn_bf, wqkvT, q_bf, k_bf, vt_bf);
  attn_core_kernel<<<dim3(B_ / S_, NSEQ / 16), 512, 0, stream>>>(
      q_bf, k_bf, vt_bf, Wpre, Wpost, mask, gates, ao_bf);
  out_gemm_kernel<<<dim3(4, ROWS / 16), 256, 0, stream>>>(ao_bf, woutT, out);
}
